// LSTMModel_84052509982988
// MI455X (gfx1250) — compile-verified, matrix operand order checked by reading, not run on hardware
//
#include <hip/hip_runtime.h>
#include <hip/hip_bf16.h>

// ---------------------------------------------------------------------------
// LSTM (B=1024, S=512, EMB=50, H=64) + FC to 50000 classes + softmax
// CDNA5 / gfx1250: wave32, WMMA f32_16x16x32_f16 (f32 accumulate).
// Design: weights register-resident across the 512-step recurrence,
// h/c state in LDS, gates exchanged cross-wave through LDS.
// Cell nonlinearities are branch-free (exp + v_rcp_f32) so they co-execute
// with WMMA as TRANS ops instead of libm tanhf's EXEC-mask branches.
// ---------------------------------------------------------------------------

#define BATCH  1024
#define SEQLEN 512
#define EMBD   50
#define HID    64
#define G4     256      // 4*HID gate columns
#define KPAD   64       // EMB padded to WMMA K granularity
#define NCITY  50000

typedef __attribute__((ext_vector_type(16))) _Float16 v16h;
typedef __attribute__((ext_vector_type(8)))  _Float16 v8h;
typedef __attribute__((ext_vector_type(8)))  float    v8f;

#define WMMA_F16(A, B, C) \
  __builtin_amdgcn_wmma_f32_16x16x32_f16(false, (A), false, (B), (short)0, (C), false, false)

// A/B fragment K mapping for 16x16x32 f16 (ISA 7.12.2):
// lane<16 holds K {0..7,16..23}, lane>=16 holds K {8..15,24..31}.
__device__ __forceinline__ v16h frag_from_lds(const _Float16* p, int lane) {
  int kb = (lane >> 4) << 3;                 // 0 or 8
  v8h lo = *(const v8h*)(p + kb);            // K = kb .. kb+7      (ds_load_b128)
  v8h hi = *(const v8h*)(p + kb + 16);       // K = kb+16 .. kb+23  (ds_load_b128)
  return __builtin_shufflevector(lo, hi, 0,1,2,3,4,5,6,7,8,9,10,11,12,13,14,15);
}

// Build a B fragment (K x N, element [k][n] = W[n][k]) from global f32 weights,
// converting to f16. Done once per wave before the recurrence loop.
__device__ __forceinline__ v16h wfrag_from_global(const float* __restrict__ W,
                                                  int n, int ldk, int kvalid,
                                                  int kc, int lane) {
  v16h f;
  int kb = (lane >> 4) << 3;
#pragma unroll
  for (int e = 0; e < 16; ++e) {
    int k = kc * 32 + (e & 7) + ((e >> 3) << 4) + kb;
    f[e] = (k < kvalid) ? (_Float16)W[n * ldk + k] : (_Float16)0.0f;
  }
  return f;
}

// Branch-free sigmoid/tanh: v_exp_f32 + v_rcp_f32 (TRANS ops, no EXEC branches).
__device__ __forceinline__ float sigmoid_fast(float x) {
  return __builtin_amdgcn_rcpf(1.0f + __expf(-x));
}
__device__ __forceinline__ float tanh_fast(float x) {
  return 2.0f * __builtin_amdgcn_rcpf(1.0f + __expf(-2.0f * x)) - 1.0f;
}

// ---------------------------------------------------------------------------
// Kernel 1: persistent LSTM. 64 blocks x 256 threads (8 waves).
// Block = 16 batch rows. Wave w owns gate columns [w*32, w*32+32).
// ---------------------------------------------------------------------------
__global__ __launch_bounds__(256)
void lstm_kernel(const int* __restrict__ x, const float* __restrict__ emb,
                 const float* __restrict__ W_ih, const float* __restrict__ W_hh,
                 const float* __restrict__ b_ih, const float* __restrict__ b_hh,
                 float* __restrict__ hOut) {
  __shared__ __align__(16) _Float16 eLds[16 * KPAD];   // x_t embeddings (f16)
  __shared__ __align__(16) _Float16 hLds[16 * HID];    // h_{t-1} (f16, WMMA feed)
  __shared__ __align__(16) float    gLds[16 * G4];     // raw gates (f32)
  __shared__ __align__(16) float    cLds[16 * HID];    // cell state (f32)
  __shared__ float biasLds[G4];

  const int tid  = threadIdx.x;
  const int lane = tid & 31;
  const int w    = tid >> 5;
  const int b0   = blockIdx.x * 16;

  // --- Preload weight B-fragments into VGPRs (resident for all 512 steps) ---
  v16h bih[2][2], bhh[2][2];                 // [n-subtile][k-chunk]
#pragma unroll
  for (int st = 0; st < 2; ++st) {
    int n = w * 32 + st * 16 + (lane & 15);
#pragma unroll
    for (int kc = 0; kc < 2; ++kc) {
      bih[st][kc] = wfrag_from_global(W_ih, n, EMBD, EMBD, kc, lane);
      bhh[st][kc] = wfrag_from_global(W_hh, n, HID,  HID,  kc, lane);
    }
  }

  // --- Init: combined bias, zero h/c ---
  biasLds[tid] = b_ih[tid] + b_hh[tid];
  const int m16 = tid >> 4;
  const int c0  = (tid & 15) * 4;
#pragma unroll
  for (int j = 0; j < 4; ++j) {
    hLds[m16 * HID + c0 + j] = (_Float16)0.0f;
    cLds[m16 * HID + c0 + j] = 0.0f;
  }
  __syncthreads();

  for (int t = 0; t < SEQLEN; ++t) {
    // --- Gather embeddings for this step into LDS (f32 -> f16, pad to 64) ---
    int idx = x[(b0 + m16) * SEQLEN + t];
#pragma unroll
    for (int j = 0; j < 4; ++j) {
      int c = c0 + j;
      eLds[m16 * KPAD + c] = (c < EMBD) ? (_Float16)emb[idx * EMBD + c]
                                        : (_Float16)0.0f;
    }
    if (t + 1 < SEQLEN) {   // warm nearest caches for next step's gather
      int idxn = x[(b0 + m16) * SEQLEN + t + 1];
      __builtin_prefetch(&emb[idxn * EMBD], 0, 3);
    }
    __syncthreads();        // B1: eLds/hLds ready; gLds consumers done

    // --- Gates = E_t @ W_ih^T + H @ W_hh^T  (8 WMMAs per wave) ---
    v8f acc0 = {}, acc1 = {};
    const int ml = lane & 15;
#pragma unroll
    for (int kc = 0; kc < 2; ++kc) {
      v16h ae = frag_from_lds(eLds + ml * KPAD + kc * 32, lane);
      acc0 = WMMA_F16(ae, bih[0][kc], acc0);
      acc1 = WMMA_F16(ae, bih[1][kc], acc1);
      v16h ah = frag_from_lds(hLds + ml * HID + kc * 32, lane);
      acc0 = WMMA_F16(ah, bhh[0][kc], acc0);
      acc1 = WMMA_F16(ah, bhh[1][kc], acc1);
    }
    {
      int mhi = (lane >> 4) << 3;
      int n0 = w * 32 + ml;
      int n1 = w * 32 + 16 + ml;
#pragma unroll
      for (int r = 0; r < 8; ++r) {
        int m = r + mhi;    // C/D layout: VGPR r -> M = r (+8 for upper lanes)
        gLds[m * G4 + n0] = acc0[r] + biasLds[n0];
        gLds[m * G4 + n1] = acc1[r] + biasLds[n1];
      }
    }
    __syncthreads();        // B2: gates ready; eLds/hLds reads done

    // --- Cell update (cross-wave: i/f/g/o live in different waves) ---
#pragma unroll
    for (int j = 0; j < 4; ++j) {
      int n = (tid & 15) + j * 16;
      float gi = sigmoid_fast(gLds[m16 * G4 + n]);
      float gf = sigmoid_fast(gLds[m16 * G4 + HID + n]);
      float gg = tanh_fast   (gLds[m16 * G4 + 2 * HID + n]);
      float go = sigmoid_fast(gLds[m16 * G4 + 3 * HID + n]);
      float c  = gf * cLds[m16 * HID + n] + gi * gg;
      cLds[m16 * HID + n] = c;
      float h  = go * tanh_fast(c);
      hLds[m16 * HID + n] = (_Float16)h;
      if (t == SEQLEN - 1) hOut[(b0 + m16) * HID + n] = h;
    }
  }
}

// ---------------------------------------------------------------------------
// Kernel 2: logits = h @ W_fc^T + b_fc. Grid (391, 64), 256 threads (8 waves).
// Block tile: M=16 batch rows x N=128 classes, K=64.
// ---------------------------------------------------------------------------
__global__ __launch_bounds__(256)
void fc_kernel(const float* __restrict__ h, const float* __restrict__ Wfc,
               const float* __restrict__ bfc, float* __restrict__ out) {
  __shared__ __align__(16) _Float16 aLds[16 * HID];
  __shared__ __align__(16) _Float16 wLds[128 * HID];

  const int tid  = threadIdx.x;
  const int lane = tid & 31;
  const int w    = tid >> 5;
  const int b0   = blockIdx.y * 16;
  const int n0   = blockIdx.x * 128;

  {
    int m = tid >> 4, cc = (tid & 15) * 4;
#pragma unroll
    for (int j = 0; j < 4; ++j)
      aLds[m * HID + cc + j] = (_Float16)h[(b0 + m) * HID + cc + j];
  }
  for (int i = tid; i < 128 * HID; i += 256) {
    int nl = i >> 6, k = i & 63;
    int n = n0 + nl;
    wLds[i] = (n < NCITY) ? (_Float16)Wfc[n * HID + k] : (_Float16)0.0f;
  }
  __syncthreads();

  v8f acc = {};
  const int ml = lane & 15;
#pragma unroll
  for (int kc = 0; kc < 2; ++kc) {
    v16h a = frag_from_lds(aLds + ml * HID + kc * 32, lane);
    v16h b = frag_from_lds(wLds + (w * 16 + ml) * HID + kc * 32, lane);
    acc = WMMA_F16(a, b, acc);
  }

  int n = n0 + w * 16 + ml;
  if (n < NCITY) {
    float bias = bfc[n];
    int mhi = (lane >> 4) << 3;
#pragma unroll
    for (int r = 0; r < 8; ++r) {
      int m = r + mhi;
      out[(size_t)(b0 + m) * NCITY + n] = acc[r] + bias;
    }
  }
}

// ---------------------------------------------------------------------------
// Kernel 3: in-place row softmax over 50000 columns. One block per row.
// ---------------------------------------------------------------------------
__global__ __launch_bounds__(256)
void softmax_kernel(float* __restrict__ out) {
  __shared__ float red[256];
  const int tid = threadIdx.x;
  float* row = out + (size_t)blockIdx.x * NCITY;

  float mx = -3.402823466e38f;
  for (int c = tid; c < NCITY; c += 256) mx = fmaxf(mx, row[c]);
  red[tid] = mx; __syncthreads();
  for (int s = 128; s > 0; s >>= 1) {
    if (tid < s) red[tid] = fmaxf(red[tid], red[tid + s]);
    __syncthreads();
  }
  mx = red[0]; __syncthreads();

  float sum = 0.0f;
  for (int c = tid; c < NCITY; c += 256) sum += __expf(row[c] - mx);
  red[tid] = sum; __syncthreads();
  for (int s = 128; s > 0; s >>= 1) {
    if (tid < s) red[tid] += red[tid + s];
    __syncthreads();
  }
  float inv = __builtin_amdgcn_rcpf(red[0]);

  for (int c = tid; c < NCITY; c += 256) row[c] = __expf(row[c] - mx) * inv;
}

// ---------------------------------------------------------------------------
extern "C" void kernel_launch(void* const* d_in, const int* in_sizes, int n_in,
                              void* d_out, int out_size, void* d_ws, size_t ws_size,
                              hipStream_t stream) {
  const int*   x   = (const int*)  d_in[0];
  const float* emb = (const float*)d_in[1];
  const float* Wih = (const float*)d_in[2];
  const float* Whh = (const float*)d_in[3];
  const float* bih = (const float*)d_in[4];
  const float* bhh = (const float*)d_in[5];
  const float* Wfc = (const float*)d_in[6];
  const float* bfc = (const float*)d_in[7];
  float* out  = (float*)d_out;
  float* hbuf = (float*)d_ws;                  // BATCH*HID f32 final hidden state

  lstm_kernel<<<BATCH / 16, 256, 0, stream>>>(x, emb, Wih, Whh, bih, bhh, hbuf);

  dim3 g((NCITY + 127) / 128, BATCH / 16);
  fc_kernel<<<g, 256, 0, stream>>>(hbuf, Wfc, bfc, out);

  softmax_kernel<<<BATCH, 256, 0, stream>>>(out);
}
